// MultiHead_54460185313730
// MI455X (gfx1250) — matrix operand order repaired and checked by reading, not compiled, not run on hardware
//
#include <hip/hip_runtime.h>
#include <hip/hip_bf16.h>
#include <cstdint>

// ---------------------------------------------------------------------------
// CDNA5 / gfx1250 bf16 WMMA building blocks (wave32)
// ---------------------------------------------------------------------------
typedef __attribute__((ext_vector_type(16))) __bf16 v16bf;
typedef __attribute__((ext_vector_type(8)))  float  v8f;

union FragU { v16bf v; uint4 q[2]; };

__device__ __forceinline__ unsigned short f2bf(float f) {
  unsigned int u = __float_as_uint(f);
  u += 0x7FFFu + ((u >> 16) & 1u);          // round-to-nearest-even
  return (unsigned short)(u >> 16);
}

__device__ __forceinline__ v8f vzero8() {
  v8f z;
#pragma unroll
  for (int i = 0; i < 8; ++i) z[i] = 0.0f;
  return z;
}

__device__ __forceinline__ v8f wmma_bf16(v16bf a, v16bf b, v8f c) {
  // (neg_a, A, neg_b, B, c_mod, C, reuse_a, reuse_b)
  return __builtin_amdgcn_wmma_f32_16x16x32_bf16(false, a, false, b, (short)0, c,
                                                 false, false);
}

// ---------------------------------------------------------------------------
// gfx1250 async global->LDS copy (16B per lane), ASYNCcnt-tracked.
// LDS address is taken from the vdst VGPR; generic-pointer low 32 bits are the LDS addr.
// ---------------------------------------------------------------------------
__device__ __forceinline__ void async_copy16(void* lds_ptr, const void* gptr) {
  unsigned lds_off = (unsigned)(uintptr_t)lds_ptr;
  unsigned long long ga = (unsigned long long)(uintptr_t)gptr;
  asm volatile("global_load_async_to_lds_b128 %0, %1, off"
               :: "v"(lds_off), "v"(ga)
               : "memory");
}

#if defined(__has_builtin) && __has_builtin(__builtin_amdgcn_s_wait_asynccnt)
#define WAIT_ASYNC(n) __builtin_amdgcn_s_wait_asynccnt(n)
#else
#define WAIT_ASYNC(n) asm volatile("s_wait_asynccnt %0" :: "i"(n) : "memory")
#endif

// A-matrix 16x32 bf16 fragment. Source laid out [row][k] (k fast), row stride ldk.
// k order used here (same as B): lane m = L&15, half = L>>4; elems e -> K = half*16 + e (16 contiguous)
__device__ __forceinline__ v16bf load_a_frag(const unsigned short* base, int ldk,
                                             int row0, int k0, int lane) {
  int m    = row0 + (lane & 15);
  int half = (lane >> 4) & 1;
  const unsigned short* p = base + (size_t)m * ldk + k0;
  FragU f;
  f.q[0] = *(const uint4*)(p + half * 16);
  f.q[1] = *(const uint4*)(p + half * 16 + 8);
  return f.v;
}

// B-matrix 32x16 bf16 fragment. Source laid out [n][k] (k fast), row stride ldk.
// k order used here (same as A): lane n = L&15, half = L>>4; elems e -> K = half*16 + e (16 contiguous)
__device__ __forceinline__ v16bf load_b_frag(const unsigned short* base, int ldk,
                                             int n0, int k0, int lane) {
  int n    = n0 + (lane & 15);
  int half = (lane >> 4) & 1;
  const unsigned short* p = base + (size_t)n * ldk + k0 + half * 16;
  FragU f;
  f.q[0] = *(const uint4*)(p);
  f.q[1] = *(const uint4*)(p + 8);
  return f.v;
}

// ---------------------------------------------------------------------------
// Kernel 1: fp32 -> bf16 conversion (vectorized x4)
// ---------------------------------------------------------------------------
__global__ __launch_bounds__(256)
void cvt_f32_bf16_kernel(const float* __restrict__ in,
                         unsigned short* __restrict__ out, int n4) {
  int i = blockIdx.x * 256 + threadIdx.x;
  if (i < n4) {
    float4 f = ((const float4*)in)[i];
    uint2 p;
    p.x = (unsigned)f2bf(f.x) | ((unsigned)f2bf(f.y) << 16);
    p.y = (unsigned)f2bf(f.z) | ((unsigned)f2bf(f.w) << 16);
    ((uint2*)out)[i] = p;
  }
}

// ---------------------------------------------------------------------------
// Kernel 2: projection GEMM, "NT" form (both operands K-fast, bf16).
//   C[m][n] = sum_k A[m][k] * B[n][k] + bias[m]   (C bf16, row-major n-fast)
// Double-buffered async global->LDS staging; each thread issues exactly
// 3 async copies per K-tile, so s_wait_asynccnt 3 retires the previous tile.
// ---------------------------------------------------------------------------
__global__ __launch_bounds__(256)
void gemm_nt_bias_bf16(const unsigned short* __restrict__ A,
                       const unsigned short* __restrict__ B,
                       const float* __restrict__ bias,
                       unsigned short* __restrict__ C,
                       int M, int N, int K) {
  __shared__ __align__(16) unsigned short sA[2][64][32];
  __shared__ __align__(16) unsigned short sB[2][128][32];

  int tid  = threadIdx.x;
  int lane = tid & 31;
  int wave = tid >> 5;
  int wm   = wave >> 2;   // 0..1
  int wn   = wave & 3;    // 0..3
  int m_blk = blockIdx.x * 64;
  int n_blk = blockIdx.y * 128;

  int arow = tid >> 2, ach = tid & 3;          // A: 64 rows x 4 chunks
  const unsigned short* gA = A + (size_t)(m_blk + arow) * K + ach * 8;
  const unsigned short* gB0 = B + (size_t)(n_blk + arow) * K + ach * 8;
  const unsigned short* gB1 = B + (size_t)(n_blk + arow + 64) * K + ach * 8;

  v8f acc[2][2];
  acc[0][0] = vzero8(); acc[0][1] = vzero8();
  acc[1][0] = vzero8(); acc[1][1] = vzero8();

  // prologue: stage tile 0 into buffer 0
  async_copy16(&sA[0][arow][ach * 8], gA);
  async_copy16(&sB[0][arow][ach * 8], gB0);
  async_copy16(&sB[0][arow + 64][ach * 8], gB1);

  const int nsteps = K >> 5;
  for (int t = 0; t < nsteps; ++t) {
    int cur = t & 1;
    if (t + 1 < nsteps) {
      int nxt = cur ^ 1, k1 = (t + 1) << 5;
      async_copy16(&sA[nxt][arow][ach * 8], gA + k1);
      async_copy16(&sB[nxt][arow][ach * 8], gB0 + k1);
      async_copy16(&sB[nxt][arow + 64][ach * 8], gB1 + k1);
      WAIT_ASYNC(3);
    } else {
      WAIT_ASYNC(0);
    }
    __syncthreads();

    v16bf a0 = load_a_frag(&sA[cur][0][0], 32, wm * 32,      0, lane);
    v16bf a1 = load_a_frag(&sA[cur][0][0], 32, wm * 32 + 16, 0, lane);
    v16bf b0 = load_b_frag(&sB[cur][0][0], 32, wn * 32,      0, lane);
    v16bf b1 = load_b_frag(&sB[cur][0][0], 32, wn * 32 + 16, 0, lane);
    acc[0][0] = wmma_bf16(a0, b0, acc[0][0]);
    acc[0][1] = wmma_bf16(a0, b1, acc[0][1]);
    acc[1][0] = wmma_bf16(a1, b0, acc[1][0]);
    acc[1][1] = wmma_bf16(a1, b1, acc[1][1]);
    __syncthreads();
  }

  int nl = lane & 15, half = lane >> 4;
#pragma unroll
  for (int im = 0; im < 2; ++im)
#pragma unroll
    for (int in2 = 0; in2 < 2; ++in2)
#pragma unroll
      for (int j = 0; j < 8; ++j) {
        int m = m_blk + wm * 32 + im * 16 + half * 8 + j;
        int n = n_blk + wn * 32 + in2 * 16 + nl;
        C[(size_t)m * N + n] = f2bf(acc[im][in2][j] + bias[m]);
      }
}

// ---------------------------------------------------------------------------
// Kernel 3: fused scores -> group-softmax(64) -> attn, one block per
// (o = h*64+dk, 32-row b-block). q/k/v are bf16 [s=256][b or c=256] matrices
// at Px + o*65536. Output hcT[o][s*256 + b] = attn[b][s] (bf16).
// ---------------------------------------------------------------------------
__global__ __launch_bounds__(256)
void attention_kernel(const unsigned short* __restrict__ Pq,
                      const unsigned short* __restrict__ Pk,
                      const unsigned short* __restrict__ Pv,
                      unsigned short* __restrict__ hcT) {
  const int o  = blockIdx.x;       // 0..1023
  const int b0 = blockIdx.y * 32;  // 0,32,...,224
  const unsigned short* q = Pq + (size_t)o * 65536;
  const unsigned short* k = Pk + (size_t)o * 65536;
  const unsigned short* v = Pv + (size_t)o * 65536;

  __shared__ __align__(16) float          S[32][256];     // scores (f32)   32KB
  __shared__ __align__(16) unsigned short soft[32][256];  // softmax (bf16) 16KB
  __shared__ __align__(16) unsigned short at[32][32];     // q^T tile [b][s] 2KB
  __shared__ __align__(16) unsigned short bt[128][32];    // k^T tile [c][s] 8KB

  int tid  = threadIdx.x;
  int lane = tid & 31;
  int wave = tid >> 5;
  int wm   = wave >> 2;   // 0..1  (16 b-rows each)
  int wn   = wave & 3;    // 0..3
  int nl   = lane & 15, half = lane >> 4;

  // ---- Phase 1: S[b][c] = (1/8) * sum_s q[s][b] * k[s][c], two 128-col halves
  for (int chalf = 0; chalf < 2; ++chalf) {
    int cbase = chalf * 128;
    v8f acc[2]; acc[0] = vzero8(); acc[1] = vzero8();

    for (int s0 = 0; s0 < 256; s0 += 32) {
      // transpose-stage q: at[bi][si] = q[(s0+si)*256 + b0+bi]  (128 tasks)
      if (tid < 128) {
        int si = tid >> 2, ch = tid & 3;
        const unsigned short* g = q + (size_t)(s0 + si) * 256 + b0 + ch * 8;
        __builtin_prefetch(g + 32 * 256, 0, 1);
        uint4 gv = *(const uint4*)g;
        const unsigned short* u = (const unsigned short*)&gv;
#pragma unroll
        for (int j = 0; j < 8; ++j) at[ch * 8 + j][si] = u[j];
      }
      // transpose-stage k: bt[ci][si] = k[(s0+si)*256 + cbase+ci] (2x256 tasks)
      {
        int si = tid >> 4, ch = tid & 15;
        const unsigned short* g = k + (size_t)(s0 + si) * 256 + cbase + ch * 8;
        uint4 gv = *(const uint4*)g;
        const unsigned short* u = (const unsigned short*)&gv;
#pragma unroll
        for (int j = 0; j < 8; ++j) bt[ch * 8 + j][si] = u[j];

        const unsigned short* g2 = g + (size_t)16 * 256;
        uint4 gv2 = *(const uint4*)g2;
        const unsigned short* u2 = (const unsigned short*)&gv2;
#pragma unroll
        for (int j = 0; j < 8; ++j) bt[ch * 8 + j][si + 16] = u2[j];
      }
      __syncthreads();

      v16bf a  = load_a_frag(&at[0][0], 32, wm * 16,      0, lane);
      v16bf bA = load_b_frag(&bt[0][0], 32, wn * 32,      0, lane);
      v16bf bB = load_b_frag(&bt[0][0], 32, wn * 32 + 16, 0, lane);
      acc[0] = wmma_bf16(a, bA, acc[0]);
      acc[1] = wmma_bf16(a, bB, acc[1]);
      __syncthreads();
    }
#pragma unroll
    for (int sub = 0; sub < 2; ++sub)
#pragma unroll
      for (int j = 0; j < 8; ++j) {
        int m = wm * 16 + half * 8 + j;
        int c = cbase + wn * 32 + sub * 16 + nl;
        S[m][c] = acc[sub][j] * 0.125f;   // 1/sqrt(dk=64)
      }
  }
  __syncthreads();

  // ---- Phase 2: softmax over flat groups of 64 along c (4 groups per row)
  if (tid < 128) {
    int b = tid >> 2, g = tid & 3;
    float* row = &S[b][g * 64];
    float mx = -1e30f;
#pragma unroll 8
    for (int i = 0; i < 64; ++i) mx = fmaxf(mx, row[i]);
    float sum = 0.0f;
#pragma unroll 8
    for (int i = 0; i < 64; ++i) sum += __expf(row[i] - mx);
    float inv = __frcp_rn(sum);
#pragma unroll 8
    for (int i = 0; i < 64; ++i)
      soft[b][g * 64 + i] = f2bf(__expf(row[i] - mx) * inv);
  }
  __syncthreads();

  // ---- Phase 3: attn[b][s] = sum_c soft[b][c] * v[s][c]
  // A-frags straight from soft (already [b][c], K-fast); B-frags straight from
  // global v (row-major [s][c], K-fast). Wave: 16(b) x 64(s), 4 accums.
  v8f acc3[4];
#pragma unroll
  for (int i = 0; i < 4; ++i) acc3[i] = vzero8();

  for (int c0 = 0; c0 < 256; c0 += 32) {
    v16bf a = load_a_frag(&soft[0][0], 256, wm * 16, c0, lane);
#pragma unroll
    for (int sub = 0; sub < 4; ++sub) {
      v16bf bf = load_b_frag(v, 256, wn * 64 + sub * 16, c0, lane);
      acc3[sub] = wmma_bf16(a, bf, acc3[sub]);
    }
  }

  unsigned short* op = hcT + (size_t)o * 65536;
#pragma unroll
  for (int sub = 0; sub < 4; ++sub)
#pragma unroll
    for (int j = 0; j < 8; ++j) {
      int s = wn * 64 + sub * 16 + nl;
      int m = wm * 16 + half * 8 + j;
      op[(size_t)s * 256 + b0 + m] = f2bf(acc3[sub][j]);
    }
}

// ---------------------------------------------------------------------------
// Kernel 4: output GEMM, A transposed in memory.
//   C[m][n] = sum_k At[k][m] * B[n][k] + bias[n]   (C fp32)
// At = hcT [K=1024 rows][M=65536], B = Wo bf16 [N=1024][K=1024].
// B staged async double-buffered; A transpose-staged via ds scatter.
// ---------------------------------------------------------------------------
__global__ __launch_bounds__(256)
void gemm_tn_bias_f32(const unsigned short* __restrict__ At,
                      const unsigned short* __restrict__ B,
                      const float* __restrict__ bias,
                      float* __restrict__ C,
                      int M, int N, int K) {
  __shared__ __align__(16) unsigned short sA[2][64][32];
  __shared__ __align__(16) unsigned short sB[2][128][32];

  int tid  = threadIdx.x;
  int lane = tid & 31;
  int wave = tid >> 5;
  int wm   = wave >> 2;
  int wn   = wave & 3;
  int m_blk = blockIdx.x * 64;
  int n_blk = blockIdx.y * 128;

  int brow = tid >> 2, bch = tid & 3;
  const unsigned short* gB0 = B + (size_t)(n_blk + brow) * K + bch * 8;
  const unsigned short* gB1 = B + (size_t)(n_blk + brow + 64) * K + bch * 8;
  int oi = tid >> 3, och = tid & 7;          // A transpose: 32 o x 8 chunks

  v8f acc[2][2];
  acc[0][0] = vzero8(); acc[0][1] = vzero8();
  acc[1][0] = vzero8(); acc[1][1] = vzero8();

  // prologue: stage tile 0
  {
    const unsigned short* g = At + (size_t)oi * M + m_blk + och * 8;
    uint4 gv = *(const uint4*)g;
    const unsigned short* u = (const unsigned short*)&gv;
#pragma unroll
    for (int j = 0; j < 8; ++j) sA[0][och * 8 + j][oi] = u[j];
  }
  async_copy16(&sB[0][brow][bch * 8], gB0);
  async_copy16(&sB[0][brow + 64][bch * 8], gB1);

  const int nsteps = K >> 5;
  for (int t = 0; t < nsteps; ++t) {
    int cur = t & 1;
    if (t + 1 < nsteps) {
      int nxt = cur ^ 1, k1 = (t + 1) << 5;
      const unsigned short* g = At + (size_t)(k1 + oi) * M + m_blk + och * 8;
      __builtin_prefetch(g + (size_t)32 * M, 0, 1);
      uint4 gv = *(const uint4*)g;
      const unsigned short* u = (const unsigned short*)&gv;
#pragma unroll
      for (int j = 0; j < 8; ++j) sA[nxt][och * 8 + j][oi] = u[j];
      async_copy16(&sB[nxt][brow][bch * 8], gB0 + k1);
      async_copy16(&sB[nxt][brow + 64][bch * 8], gB1 + k1);
      WAIT_ASYNC(2);
    } else {
      WAIT_ASYNC(0);
    }
    __syncthreads();

    v16bf a0 = load_a_frag(&sA[cur][0][0], 32, wm * 32,      0, lane);
    v16bf a1 = load_a_frag(&sA[cur][0][0], 32, wm * 32 + 16, 0, lane);
    v16bf b0 = load_b_frag(&sB[cur][0][0], 32, wn * 32,      0, lane);
    v16bf b1 = load_b_frag(&sB[cur][0][0], 32, wn * 32 + 16, 0, lane);
    acc[0][0] = wmma_bf16(a0, b0, acc[0][0]);
    acc[0][1] = wmma_bf16(a0, b1, acc[0][1]);
    acc[1][0] = wmma_bf16(a1, b0, acc[1][0]);
    acc[1][1] = wmma_bf16(a1, b1, acc[1][1]);
    __syncthreads();
  }

  int nl = lane & 15, half = lane >> 4;
#pragma unroll
  for (int im = 0; im < 2; ++im)
#pragma unroll
    for (int in2 = 0; in2 < 2; ++in2)
#pragma unroll
      for (int j = 0; j < 8; ++j) {
        int m = m_blk + wm * 32 + im * 16 + half * 8 + j;
        int n = n_blk + wn * 32 + in2 * 16 + nl;
        C[(size_t)m * N + n] = acc[im][in2][j] + bias[n];
      }
}

// ---------------------------------------------------------------------------
// Host-side launch
// ---------------------------------------------------------------------------
extern "C" void kernel_launch(void* const* d_in, const int* in_sizes, int n_in,
                              void* d_out, int out_size, void* d_ws, size_t ws_size,
                              hipStream_t stream) {
  const float* Q  = (const float*)d_in[0];
  const float* K  = (const float*)d_in[1];
  const float* V  = (const float*)d_in[2];
  const float* Wq = (const float*)d_in[3];
  const float* bq = (const float*)d_in[4];
  const float* Wk = (const float*)d_in[5];
  const float* bk = (const float*)d_in[6];
  const float* Wv = (const float*)d_in[7];
  const float* bv = (const float*)d_in[8];
  const float* Wo = (const float*)d_in[9];
  const float* bo = (const float*)d_in[10];
  float* out = (float*)d_out;

  const int DM = 1024;      // d_model = H*DK
  const int R  = 65536;     // N0*N1 flattened rows

  char* ws = (char*)d_ws;
  size_t off = 0;
  auto alloc = [&](size_t bytes) {
    char* p = ws + off;
    off += (bytes + 255) & ~(size_t)255;
    return p;
  };
  unsigned short* Qb  = (unsigned short*)alloc((size_t)R * DM * 2);
  unsigned short* Kb  = (unsigned short*)alloc((size_t)R * DM * 2);
  unsigned short* Vb  = (unsigned short*)alloc((size_t)R * DM * 2);
  unsigned short* Wqb = (unsigned short*)alloc((size_t)DM * DM * 2);
  unsigned short* Wkb = (unsigned short*)alloc((size_t)DM * DM * 2);
  unsigned short* Wvb = (unsigned short*)alloc((size_t)DM * DM * 2);
  unsigned short* Wob = (unsigned short*)alloc((size_t)DM * DM * 2);
  unsigned short* Pq  = (unsigned short*)alloc((size_t)DM * R * 2);
  unsigned short* Pk  = (unsigned short*)alloc((size_t)DM * R * 2);
  unsigned short* Pv  = (unsigned short*)alloc((size_t)DM * R * 2);
  unsigned short* hcT = Qb;  // Q bf16 no longer needed after its projection

  // 1) fp32 -> bf16
  int n4 = R * DM / 4;
  cvt_f32_bf16_kernel<<<(n4 + 255) / 256, 256, 0, stream>>>(Q, Qb, n4);
  cvt_f32_bf16_kernel<<<(n4 + 255) / 256, 256, 0, stream>>>(K, Kb, n4);
  cvt_f32_bf16_kernel<<<(n4 + 255) / 256, 256, 0, stream>>>(V, Vb, n4);
  int w4 = DM * DM / 4;
  cvt_f32_bf16_kernel<<<(w4 + 255) / 256, 256, 0, stream>>>(Wq, Wqb, w4);
  cvt_f32_bf16_kernel<<<(w4 + 255) / 256, 256, 0, stream>>>(Wk, Wkb, w4);
  cvt_f32_bf16_kernel<<<(w4 + 255) / 256, 256, 0, stream>>>(Wv, Wvb, w4);
  cvt_f32_bf16_kernel<<<(w4 + 255) / 256, 256, 0, stream>>>(Wo, Wob, w4);

  // 2) projections: P[o][r] = W[o,:]·X[r,:] + b[o]
  dim3 gP(DM / 64, R / 128);
  gemm_nt_bias_bf16<<<gP, 256, 0, stream>>>(Wqb, Qb, bq, Pq, DM, R, DM);
  gemm_nt_bias_bf16<<<gP, 256, 0, stream>>>(Wkb, Kb, bk, Pk, DM, R, DM);
  gemm_nt_bias_bf16<<<gP, 256, 0, stream>>>(Wvb, Vb, bv, Pv, DM, R, DM);

  // 3) fused scores -> group-softmax(64) -> attn, per (o, b-block)
  attention_kernel<<<dim3(1024, 8), 256, 0, stream>>>(Pq, Pk, Pv, hcT);

  // 4) output projection: out[r][e] = sum_o hcT[o][r]*Wo[e][o] + bo[e]
  dim3 gO(R / 64, DM / 128);
  gemm_tn_bias_f32<<<gO, 256, 0, stream>>>(hcT, Wob, bo, out, R, DM, DM);
}
